// CasualSelfAttention_75634374082868
// MI455X (gfx1250) — matrix operand order repaired and checked by reading, not compiled, not run on hardware
//
#include <hip/hip_runtime.h>
#include <hip/hip_bf16.h>

// ---------- CDNA5 WMMA types ----------
typedef __attribute__((ext_vector_type(16))) __bf16 bf16x16;
typedef __attribute__((ext_vector_type(8)))  float  v8f;

union ABReg { unsigned int u[8]; bf16x16 v; };

__device__ __forceinline__ unsigned short f32_to_bf16u(float f) {
  unsigned int u = __float_as_uint(f);
  u += 0x7FFFu + ((u >> 16) & 1u);           // round-to-nearest-even
  return (unsigned short)(u >> 16);
}
__device__ __forceinline__ unsigned int pack_bf16(float lo, float hi) {
  return (unsigned int)f32_to_bf16u(lo) | ((unsigned int)f32_to_bf16u(hi) << 16);
}
__device__ __forceinline__ v8f zero8() {
  v8f z;
#pragma unroll
  for (int r = 0; r < 8; ++r) z[r] = 0.0f;
  return z;
}
__device__ __forceinline__ v8f wmma_bf16(bf16x16 a, bf16x16 b, v8f c) {
  // D = A(16x32 bf16) x B(32x16 bf16) + C(16x16 f32)
  return __builtin_amdgcn_wmma_f32_16x16x32_bf16(false, a, false, b, (short)0, c, false, false);
}

// ---------- operand builders (ISA 7.12.2 layouts, wave32) ----------
// A 16x32 bf16: lanes 0-15 -> M=lane, K = {8h+2j, 8h+2j+1} (v0-3), {16+8h+2j,...} (v4-7), h=lane>>4
__device__ __forceinline__ bf16x16 load_a_bf16(const unsigned short* A, int lda, int row0, int kofs, int lane) {
  int m = lane & 15, h = lane >> 4;
  const unsigned int* p = (const unsigned int*)(A + (size_t)(row0 + m) * lda + kofs + 8 * h);
  ABReg r;
#pragma unroll
  for (int j = 0; j < 4; ++j) r.u[j] = p[j];
#pragma unroll
  for (int j = 0; j < 4; ++j) r.u[4 + j] = p[8 + j];
  return r.v;
}
// A from LDS stored as packed-bf16 uints, row stride lda2 (uints)
__device__ __forceinline__ bf16x16 load_a_lds(const unsigned int* S, int lda2, int kofs, int lane) {
  int m = lane & 15, h = lane >> 4;
  const unsigned int* p = S + m * lda2 + (kofs >> 1) + 4 * h;
  ABReg r;
#pragma unroll
  for (int j = 0; j < 4; ++j) r.u[j] = p[j];
#pragma unroll
  for (int j = 0; j < 4; ++j) r.u[4 + j] = p[8 + j];
  return r.v;
}
// B 32x16 bf16: lane = N (lane&15); K = {8h+2j, 8h+2j+1} (v0-3), {16+8h+2j,...} (v4-7), h=lane>>4
// source layout: element (k,n) = src[(n0+n)*ldb + kofs + k]   (column n contiguous in k)
__device__ __forceinline__ bf16x16 load_b_bf16(const unsigned short* Bsrc, int ldb, int n0, int kofs, int lane) {
  int n = lane & 15, h = lane >> 4;
  const unsigned int* p = (const unsigned int*)(Bsrc + (size_t)(n0 + n) * ldb + kofs + 8 * h);
  ABReg r;
#pragma unroll
  for (int j = 0; j < 4; ++j) r.u[j] = p[j];
#pragma unroll
  for (int j = 0; j < 4; ++j) r.u[4 + j] = p[8 + j];
  return r.v;
}

// ---------- one-shot f32 -> bf16 conversion (8 elems / thread) ----------
__global__ void __launch_bounds__(256) cvt_bf16_kernel(const float* __restrict__ src,
                                                       unsigned int* __restrict__ dst, int n8) {
  int i = blockIdx.x * 256 + threadIdx.x;
  if (i >= n8) return;
  const float4* s = (const float4*)src + 2 * (size_t)i;
  float4 a = s[0], b = s[1];
  uint4 o;
  o.x = pack_bf16(a.x, a.y);
  o.y = pack_bf16(a.z, a.w);
  o.z = pack_bf16(b.x, b.y);
  o.w = pack_bf16(b.z, b.w);
  ((uint4*)dst)[i] = o;
}

// ---------- GEMM: out[M=16384, N=768] = A[16384,768] @ W[768,768]^T + bias ----------
// All-bf16 operands. Each wave: 16x128 output tile (8 WMMAs per A load).
// OUT_MODE 0: bf16 row-major; 1: bf16 per-batch transposed [b][n][s]; 2: f32 row-major
template <int OUT_MODE>
__global__ void __launch_bounds__(256) gemm768_kernel(const unsigned short* __restrict__ A,
                                                      const unsigned short* __restrict__ W,
                                                      const float* __restrict__ bias,
                                                      void* __restrict__ Out) {
  const int K = 768, N = 768;
  int tid = threadIdx.x, w = tid >> 5, lane = tid & 31;
  int waveTile = blockIdx.x * 8 + w;          // 1024 * 6 wave tiles (each 16x128)
  int mt = waveTile / 6;
  int n0 = (waveTile % 6) * 128;

  v8f acc[8];
#pragma unroll
  for (int t = 0; t < 8; ++t) acc[t] = zero8();
  for (int kofs = 0; kofs < K; kofs += 32) {
    bf16x16 a = load_a_bf16(A, K, mt * 16, kofs, lane);
#pragma unroll
    for (int t = 0; t < 8; ++t) {
      bf16x16 b = load_b_bf16(W, K, n0 + 16 * t, kofs, lane);
      acc[t] = wmma_bf16(a, b, acc[t]);
    }
  }
  int col = lane & 15, h = lane >> 4;
#pragma unroll
  for (int t = 0; t < 8; ++t) {
    int nn = n0 + 16 * t + col;
    float bv = bias[nn];
#pragma unroll
    for (int r = 0; r < 8; ++r) {
      int row = mt * 16 + r + 8 * h;
      float val = acc[t][r] + bv;
      if constexpr (OUT_MODE == 0) {
        ((unsigned short*)Out)[(size_t)row * N + nn] = f32_to_bf16u(val);
      } else if constexpr (OUT_MODE == 1) {
        int bb = row >> 12, s = row & 4095;   // 4096 rows per batch
        ((unsigned short*)Out)[(size_t)bb * N * 4096 + (size_t)nn * 4096 + s] = f32_to_bf16u(val);
      } else {
        ((float*)Out)[(size_t)row * N + nn] = val;
      }
    }
  }
}

// ---------- Flash attention: softmax(Q K^T * 768^-0.5) V, output transposed bf16 ----------
// 1 workgroup (8 waves) per (batch, 16-query block). Wave w owns dims [96w, 96w+96).
__global__ void __launch_bounds__(256) flash_attn_kernel(const unsigned short* __restrict__ Q,
                                                         const unsigned short* __restrict__ Kmat,
                                                         const unsigned short* __restrict__ Vt,
                                                         unsigned short* __restrict__ Yt) {
  const int S = 4096, D = 768;
  __shared__ unsigned int sQ[16 * 384];        // 16 x 768 bf16 (packed pairs)
  __shared__ float        sPart[8][16 * 32];   // per-wave partial logits
  __shared__ float        sSsum[16 * 32];      // reduced logits
  __shared__ unsigned int sP[16 * 16];         // 16 x 32 probabilities, bf16 packed
  __shared__ float        sM[16], sL[16], sAlpha[16];

  int tid = threadIdx.x, w = tid >> 5, lane = tid & 31;
  int qb = blockIdx.x;                 // 0..1023
  int bi = qb >> 8;                    // 256 query blocks per batch
  int q0 = (qb & 255) * 16;            // query row within batch
  const unsigned short* Qb = Q    + (size_t)bi * S * D;
  const unsigned short* Kb = Kmat + (size_t)bi * S * D;
  const unsigned short* Vb = Vt   + (size_t)bi * D * S;
  unsigned short*       Yb = Yt   + (size_t)bi * D * S;

  { // stage Q block into LDS
    const unsigned int* qg = (const unsigned int*)(Qb + (size_t)q0 * D);
    for (int i = tid; i < 16 * 384; i += 256) sQ[i] = qg[i];
  }
  if (tid < 16) { sM[tid] = -1e30f; sL[tid] = 0.0f; }
  __syncthreads();

  v8f o[6];
#pragma unroll
  for (int t = 0; t < 6; ++t) o[t] = zero8();
  const int kslice = w * 96;
  const float scale = 0.036084391824351615f;   // 768^-0.5

  for (int kb = 0; kb < 128; ++kb) {
    int key0 = kb * 32;
    if (kb + 1 < 128)
      __builtin_prefetch(Kb + (size_t)(key0 + 32) * D, 0, 1);

    // partial logits over this wave's 96-dim slice
    v8f s0 = zero8(), s1 = zero8();
#pragma unroll
    for (int kk = 0; kk < 3; ++kk) {
      int kofs = kslice + kk * 32;
      bf16x16 a  = load_a_lds(sQ, 384, kofs, lane);
      bf16x16 b0 = load_b_bf16(Kb, D, key0,      kofs, lane);
      bf16x16 b1 = load_b_bf16(Kb, D, key0 + 16, kofs, lane);
      s0 = wmma_bf16(a, b0, s0);
      s1 = wmma_bf16(a, b1, s1);
    }
    { // dump partials to LDS (C layout: row = r + 8h, col = lane&15)
      int col = lane & 15, h = lane >> 4;
      float* dst = &sPart[w][0];
#pragma unroll
      for (int r = 0; r < 8; ++r) {
        int row = r + 8 * h;
        dst[row * 32 + col]      = s0[r];
        dst[row * 32 + col + 16] = s1[r];
      }
    }
    __syncthreads();
    // cross-wave reduction + scale
    for (int e = tid; e < 512; e += 256) {
      float acc = 0.0f;
#pragma unroll
      for (int ww = 0; ww < 8; ++ww) acc += sPart[ww][e];
      sSsum[e] = acc * scale;
    }
    __syncthreads();
    // online softmax per row (16 rows handled by threads 0..15)
    if (tid < 16) {
      int row = tid;
      float mold = sM[row], mx = mold;
      float sv[32];
#pragma unroll
      for (int j = 0; j < 32; ++j) { sv[j] = sSsum[row * 32 + j]; mx = fmaxf(mx, sv[j]); }
      float alpha = __expf(mold - mx);
      float lsum = 0.0f;
#pragma unroll
      for (int j = 0; j < 32; ++j) { float p = __expf(sv[j] - mx); sv[j] = p; lsum += p; }
      sL[row] = sL[row] * alpha + lsum;
      sM[row] = mx;
      sAlpha[row] = alpha;
#pragma unroll
      for (int j = 0; j < 16; ++j) sP[row * 16 + j] = pack_bf16(sv[2 * j], sv[2 * j + 1]);
    }
    __syncthreads();
    { // rescale accumulators by alpha[row]
      int h = lane >> 4;
      float al[8];
#pragma unroll
      for (int r = 0; r < 8; ++r) al[r] = sAlpha[r + 8 * h];
#pragma unroll
      for (int t = 0; t < 6; ++t)
#pragma unroll
        for (int r = 0; r < 8; ++r) o[t][r] *= al[r];
    }
    // O += P(16x32) @ V(32 x 96-slice); Vt layout [d][s] -> column-contiguous B loads
    bf16x16 ap = load_a_lds(sP, 16, 0, lane);
#pragma unroll
    for (int t = 0; t < 6; ++t) {
      bf16x16 bv = load_b_bf16(Vb, S, kslice + 16 * t, key0, lane);
      o[t] = wmma_bf16(ap, bv, o[t]);
    }
    // (no trailing barrier needed: next-iteration LDS writes sit behind 2 barriers)
  }

  { // epilogue: divide by row sums, store transposed bf16 (this IS the reference's permutation)
    int col = lane & 15, h = lane >> 4;
    float linv[8];
#pragma unroll
    for (int r = 0; r < 8; ++r) { float l = sL[r + 8 * h]; linv[r] = (l > 0.0f) ? 1.0f / l : 0.0f; }
#pragma unroll
    for (int t = 0; t < 6; ++t) {
      int d = kslice + 16 * t + col;
#pragma unroll
      for (int r = 0; r < 8; ++r) {
        int srow = q0 + r + 8 * h;
        Yb[(size_t)d * S + srow] = f32_to_bf16u(o[t][r] * linv[r]);
      }
    }
  }
}

// ---------- host launch ----------
extern "C" void kernel_launch(void* const* d_in, const int* in_sizes, int n_in,
                              void* d_out, int out_size, void* d_ws, size_t ws_size,
                              hipStream_t stream) {
  (void)in_sizes; (void)n_in; (void)out_size; (void)ws_size;
  const float* x  = (const float*)d_in[0];
  const float* Wq = (const float*)d_in[1];
  const float* bq = (const float*)d_in[2];
  const float* Wk = (const float*)d_in[3];
  const float* bk = (const float*)d_in[4];
  const float* Wv = (const float*)d_in[5];
  const float* bv = (const float*)d_in[6];
  const float* Wc = (const float*)d_in[7];
  const float* bc = (const float*)d_in[8];
  float* out = (float*)d_out;

  const size_t ELT = (size_t)16384 * 768;      // B*S*D elements
  const size_t WE  = (size_t)768 * 768;        // weight elements
  unsigned short* Q   = (unsigned short*)d_ws;
  unsigned short* Kk  = Q   + ELT;
  unsigned short* Vt  = Kk  + ELT;
  unsigned short* Yt  = Vt  + ELT;
  unsigned short* xb  = Yt  + ELT;
  unsigned short* Wqb = xb  + ELT;
  unsigned short* Wkb = Wqb + WE;
  unsigned short* Wvb = Wkb + WE;
  unsigned short* Wcb = Wvb + WE;

  // one-shot f32 -> bf16 conversions (x read once instead of 3x in f32)
  const int XN8 = (int)(ELT / 8), WN8 = (int)(WE / 8);
  cvt_bf16_kernel<<<(XN8 + 255) / 256, 256, 0, stream>>>(x,  (unsigned int*)xb,  XN8);
  cvt_bf16_kernel<<<(WN8 + 255) / 256, 256, 0, stream>>>(Wq, (unsigned int*)Wqb, WN8);
  cvt_bf16_kernel<<<(WN8 + 255) / 256, 256, 0, stream>>>(Wk, (unsigned int*)Wkb, WN8);
  cvt_bf16_kernel<<<(WN8 + 255) / 256, 256, 0, stream>>>(Wv, (unsigned int*)Wvb, WN8);
  cvt_bf16_kernel<<<(WN8 + 255) / 256, 256, 0, stream>>>(Wc, (unsigned int*)Wcb, WN8);

  const int GEMM_BLOCKS = 768;                 // 1024*6 wave tiles / 8 waves
  gemm768_kernel<0><<<GEMM_BLOCKS, 256, 0, stream>>>(xb, Wqb, bq, Q);
  gemm768_kernel<0><<<GEMM_BLOCKS, 256, 0, stream>>>(xb, Wkb, bk, Kk);
  gemm768_kernel<1><<<GEMM_BLOCKS, 256, 0, stream>>>(xb, Wvb, bv, Vt);        // V stored [b][d][s]
  flash_attn_kernel<<<1024, 256, 0, stream>>>(Q, Kk, Vt, Yt);                 // Yt = y^T per batch
  gemm768_kernel<2><<<GEMM_BLOCKS, 256, 0, stream>>>(Yt, Wcb, bc, out);       // reads Yt row-major = permuted y
}
